// SlotAttention_56607668961395
// MI455X (gfx1250) — matrix operand order repaired and checked by reading, not compiled, not run on hardware
//
#include <hip/hip_runtime.h>
#include <hip/hip_bf16.h>

typedef __attribute__((ext_vector_type(16))) _Float16 v16h;
typedef __attribute__((ext_vector_type(8)))  float    v8f;
typedef _Float16 f16;

#define WMMA_F16(a, b, c) \
  __builtin_amdgcn_wmma_f32_16x16x32_f16(false, (a), false, (b), (short)0, (c), false, false)

union V16U { v16h h; uint4 q[2]; unsigned u[8]; };
union H8U  { f16 h[8]; uint4 q; };

static __device__ __forceinline__ float shflx(float v, int m) {
  return __shfl_xor(v, m, 32);
}
static __device__ __forceinline__ unsigned shflxu(unsigned v, int m) {
  return (unsigned)__shfl_xor((int)v, m, 32);
}

// ---------------------------------------------------------------- weights f32 -> f16
__global__ void k_convert(const float* __restrict__ a, const float* __restrict__ b,
                          const float* __restrict__ c,
                          f16* __restrict__ oa, f16* __restrict__ ob, f16* __restrict__ oc) {
  int i = blockIdx.x * blockDim.x + threadIdx.x;
  if (i < 4096) { oa[i] = (f16)a[i]; ob[i] = (f16)b[i]; oc[i] = (f16)c[i]; }
}

// ---------------------------------------------------------------- slots = mu + sigma*noise
__global__ void k_init_slots(const float* __restrict__ mu, const float* __restrict__ sigma,
                             const float* __restrict__ noise, float* __restrict__ slots) {
  int i = blockIdx.x * blockDim.x + threadIdx.x;
  if (i < 64 * 8 * 64) { int d = i & 63; slots[i] = mu[d] + sigma[d] * noise[i]; }
}

// ---------------------------------------------------------------- fused LN + K/V projection
// one wave per 16-row n-tile; A = LN(x) tile (f16), B = weight (f16), C f32.
__global__ __launch_bounds__(256) void k_ln_proj(
    const float* __restrict__ x, const f16* __restrict__ wk, const f16* __restrict__ wv,
    const float* __restrict__ bk, const float* __restrict__ bv,
    f16* __restrict__ k_h, f16* __restrict__ v_t) {
  int tid = threadIdx.x;
  int l = tid & 15, hi = (tid >> 4) & 1, wave = tid >> 5;
  int T = blockIdx.x * 8 + wave;
  int b = T >> 8, n0 = (T & 255) << 4;

  const float* xp = x + ((size_t)(b * 4096 + n0 + l)) * 64 + hi * 32;
  float xr[32];
#pragma unroll
  for (int i = 0; i < 8; ++i) {
    float4 t = *(const float4*)(xp + i * 4);
    xr[i*4+0] = t.x; xr[i*4+1] = t.y; xr[i*4+2] = t.z; xr[i*4+3] = t.w;
  }
  float s = 0.f;
#pragma unroll
  for (int i = 0; i < 32; ++i) s += xr[i];
  s += shflx(s, 16);
  float mean = s * (1.0f / 64.0f);
  float v2 = 0.f;
#pragma unroll
  for (int i = 0; i < 32; ++i) { float d = xr[i] - mean; v2 += d * d; }
  v2 += shflx(v2, 16);
  float rstd = rsqrtf(v2 * (1.0f / 64.0f) + 1e-5f);

  // build both halves of the A tiles; patterns P0={0..15}, P1={16..31}.
  V16U c0, c1;
#pragma unroll
  for (int j = 0; j < 16; ++j) {
    int i0 = j;                         // P0 local indices {0-15}
    int i1 = j + 16;                    // P1 local indices {16-31}
    c0.h[j] = (f16)((xr[i0] - mean) * rstd);
    c1.h[j] = (f16)((xr[i1] - mean) * rstd);
  }
  V16U keep = hi ? c1 : c0;
  V16U give = hi ? c0 : c1;
#pragma unroll
  for (int u = 0; u < 8; ++u) give.u[u] = shflxu(give.u[u], 16);
  v16h a0 = hi ? give.h : keep.h;  // K-step 0 (d 0..31)
  v16h a1 = hi ? keep.h : give.h;  // K-step 1 (d 32..63)

  size_t krow_base = (size_t)(b * 4096 + n0) * 64;
  size_t vrow_base = (size_t)b * 64 * 4096;
#pragma unroll
  for (int t = 0; t < 4; ++t) {
    int o = 16 * t + l;
    {   // K projection -> row-major k_h[b][n][64]
      v16h b0 = *(const v16h*)(wk + o * 64 + 16 * hi);
      v16h b1 = *(const v16h*)(wk + o * 64 + 32 + 16 * hi);
      v8f c = {};
      c = WMMA_F16(a0, b0, c);
      c = WMMA_F16(a1, b1, c);
      float bias = bk[o];
#pragma unroll
      for (int r = 0; r < 8; ++r)
        k_h[krow_base + (size_t)(r + 8 * hi) * 64 + o] = (f16)(c[r] + bias);
    }
    {   // V projection -> transposed v_t[b][d][n] (contiguous 16B stores)
      v16h b0 = *(const v16h*)(wv + o * 64 + 16 * hi);
      v16h b1 = *(const v16h*)(wv + o * 64 + 32 + 16 * hi);
      v8f c = {};
      c = WMMA_F16(a0, b0, c);
      c = WMMA_F16(a1, b1, c);
      float bias = bv[o];
      H8U p;
#pragma unroll
      for (int r = 0; r < 8; ++r) p.h[r] = (f16)(c[r] + bias);
      *(uint4*)(v_t + vrow_base + (size_t)o * 4096 + n0 + 8 * hi) = p.q;
    }
  }
}

// ---------------------------------------------------------------- q = LN(slots) @ WqT + b
// also zeroes per-iteration reduction buffers. one wave per batch.
__global__ __launch_bounds__(32) void k_q(
    const float* __restrict__ slots, const f16* __restrict__ wq, const float* __restrict__ bq,
    f16* __restrict__ q_h, float* __restrict__ colsum, float* __restrict__ upd_acc) {
  int b = blockIdx.x;
  int tid = threadIdx.x;
  int l = tid & 15, hi = tid >> 4;
  if (tid < 16) colsum[b * 16 + tid] = 0.f;
  for (int i = tid; i < 1024; i += 32) upd_acc[(size_t)b * 1024 + i] = 0.f;

  const float* sr = slots + ((size_t)b * 8 + (l & 7)) * 64;
  float xr[64];
#pragma unroll
  for (int i = 0; i < 16; ++i) {
    float4 t = *(const float4*)(sr + i * 4);
    xr[i*4+0] = t.x; xr[i*4+1] = t.y; xr[i*4+2] = t.z; xr[i*4+3] = t.w;
  }
  float s = 0.f;
#pragma unroll
  for (int i = 0; i < 64; ++i) s += xr[i];
  float mean = s * (1.f / 64.f);
  float v2 = 0.f;
#pragma unroll
  for (int i = 0; i < 64; ++i) { float d = xr[i] - mean; v2 += d * d; }
  float rstd = rsqrtf(v2 * (1.f / 64.f) + 1e-5f);

  bool valid = (l < 8);
  v16h a0, a1;
#pragma unroll
  for (int j = 0; j < 16; ++j) {
    int d = j + 16 * hi;
    a0[j] = valid ? (f16)((xr[d] - mean) * rstd) : (f16)0.f;
    a1[j] = valid ? (f16)((xr[d + 32] - mean) * rstd) : (f16)0.f;
  }
#pragma unroll
  for (int t = 0; t < 4; ++t) {
    int o = 16 * t + l;
    v16h b0 = *(const v16h*)(wq + o * 64 + 16 * hi);
    v16h b1 = *(const v16h*)(wq + o * 64 + 32 + 16 * hi);
    v8f c = {};
    c = WMMA_F16(a0, b0, c);
    c = WMMA_F16(a1, b1, c);
    float bias = bq[o];
    if (hi == 0) {
#pragma unroll
      for (int r = 0; r < 8; ++r)
        q_h[((size_t)b * 16 + r) * 64 + o] = (f16)(c[r] + bias);
    }
  }
}

// ---------------------------------------------------------------- attn logits + softmax(+eps)
// C[M=16 n-rows, N=slots]; softmax across the 8 slot lanes; colsum via atomics;
// stores attn transposed: attn_t[b][slot][n] f16.
__global__ __launch_bounds__(256) void k_attn(
    const f16* __restrict__ k_h, const f16* __restrict__ q_h,
    f16* __restrict__ attn_t, float* __restrict__ colsum) {
  int tid = threadIdx.x;
  int l = tid & 15, hi = (tid >> 4) & 1, wave = tid >> 5;
  int T = blockIdx.x * 8 + wave;
  int b = T >> 8, n0 = (T & 255) << 4;

  const f16* krow = k_h + ((size_t)(b * 4096 + n0 + l)) * 64;
  V16U A0, A1;
  A0.q[0] = *(const uint4*)(krow + 16 * hi);
  A0.q[1] = *(const uint4*)(krow + 8 + 16 * hi);
  A1.q[0] = *(const uint4*)(krow + 32 + 16 * hi);
  A1.q[1] = *(const uint4*)(krow + 40 + 16 * hi);
  const f16* qrow = q_h + ((size_t)b * 16 + l) * 64;
  v16h B0 = *(const v16h*)(qrow + 16 * hi);
  v16h B1 = *(const v16h*)(qrow + 32 + 16 * hi);
  v8f c = {};
  c = WMMA_F16(A0.h, B0, c);
  c = WMMA_F16(A1.h, B1, c);

  float p[8]; float csum = 0.f;
#pragma unroll
  for (int r = 0; r < 8; ++r) {
    float xv = c[r] * 0.125f;                       // 1/sqrt(64)
    float m = xv;
    m = fmaxf(m, shflx(m, 1)); m = fmaxf(m, shflx(m, 2)); m = fmaxf(m, shflx(m, 4));
    float e = __expf(xv - m);
    float ss = e;
    ss += shflx(ss, 1); ss += shflx(ss, 2); ss += shflx(ss, 4);
    p[r] = e / ss + 1e-8f;
    csum += p[r];
  }
  csum += shflx(csum, 16);   // combine the two n half-tiles for this slot
  if (l < 8) {
    if (hi == 0) atomicAdd(&colsum[b * 16 + l], csum);
    H8U P;
#pragma unroll
    for (int r = 0; r < 8; ++r) P.h[r] = (f16)p[r];
    *(uint4*)(attn_t + ((size_t)(b * 16 + l)) * 4096 + n0 + 8 * hi) = P.q;
  }
}

// ---------------------------------------------------------------- updates^T = v_t @ attn_t^T
// per-wave 512-n chunk, 4 d-tiles, f32 atomic accumulation into upd_acc[b][d][slot].
__global__ __launch_bounds__(256) void k_updates(
    const f16* __restrict__ v_t, const f16* __restrict__ attn_t,
    float* __restrict__ upd_acc) {
  int tid = threadIdx.x;
  int l = tid & 15, hi = (tid >> 4) & 1, wave = tid >> 5;
  int b = blockIdx.x;
  int nbase = wave * 512;

  v8f c[4] = {v8f{}, v8f{}, v8f{}, v8f{}};
  const f16* arow = attn_t + ((size_t)(b * 16 + l)) * 4096;
  const f16* vb = v_t + (size_t)b * 64 * 4096;
  for (int k0 = nbase; k0 < nbase + 512; k0 += 32) {
    v16h B = *(const v16h*)(arow + k0 + 16 * hi);
    V16U A;
#pragma unroll
    for (int t = 0; t < 4; ++t) {
      const f16* vrow = vb + (size_t)(16 * t + l) * 4096;
      A.q[0] = *(const uint4*)(vrow + k0 + 16 * hi);
      A.q[1] = *(const uint4*)(vrow + k0 + 8 + 16 * hi);
      c[t] = WMMA_F16(A.h, B, c[t]);
    }
  }
  if (l < 8) {
    float* acc = upd_acc + (size_t)b * 1024;
#pragma unroll
    for (int t = 0; t < 4; ++t)
#pragma unroll
      for (int r = 0; r < 8; ++r)
        atomicAdd(acc + (16 * t + r + 8 * hi) * 16 + l, c[t][r]);
  }
}

// ---------------------------------------------------------------- GRU + LN + MLP (tiny, f32)
__global__ __launch_bounds__(256) void k_gru(
    const float* __restrict__ upd_acc, const float* __restrict__ colsum,
    const float* __restrict__ w_ih, const float* __restrict__ w_hh,
    const float* __restrict__ b_ih, const float* __restrict__ b_hh,
    const float* __restrict__ w1, const float* __restrict__ b1,
    const float* __restrict__ w2, const float* __restrict__ b2,
    float* __restrict__ slots, float* __restrict__ out, int write_out) {
  __shared__ float s_sl[512], s_u[512], s_gi[1536], s_gh[1536], s_act[1024], s_st[16];
  int tid = threadIdx.x;
  int b = blockIdx.x;
  for (int idx = tid; idx < 512; idx += 256) {
    int s = idx >> 6, d = idx & 63;
    s_u[idx]  = upd_acc[(size_t)b * 1024 + d * 16 + s] / colsum[b * 16 + s];
    s_sl[idx] = slots[(size_t)b * 512 + idx];
  }
  __syncthreads();
  for (int o = tid; o < 1536; o += 256) {
    int s = o / 192, j = o % 192;
    const float* wi = w_ih + j * 64;
    const float* wh = w_hh + j * 64;
    float gi = b_ih[j], gh = b_hh[j];
#pragma unroll 8
    for (int i = 0; i < 64; ++i) { gi += s_u[s*64+i] * wi[i]; gh += s_sl[s*64+i] * wh[i]; }
    s_gi[o] = gi; s_gh[o] = gh;
  }
  __syncthreads();
  for (int idx = tid; idx < 512; idx += 256) {
    int s = idx >> 6, d = idx & 63;
    float r = 1.f / (1.f + __expf(-(s_gi[s*192+d]     + s_gh[s*192+d])));
    float z = 1.f / (1.f + __expf(-(s_gi[s*192+64+d]  + s_gh[s*192+64+d])));
    float n = tanhf(s_gi[s*192+128+d] + r * s_gh[s*192+128+d]);
    s_u[idx] = (1.f - z) * n + z * s_sl[idx];   // reuse s_u as h
  }
  __syncthreads();
  if (tid < 8) {
    float s = 0.f;
    for (int i = 0; i < 64; ++i) s += s_u[tid*64+i];
    float mean = s * (1.f / 64.f);
    float v2 = 0.f;
    for (int i = 0; i < 64; ++i) { float d = s_u[tid*64+i] - mean; v2 += d * d; }
    s_st[tid]     = mean;
    s_st[8 + tid] = rsqrtf(v2 * (1.f / 64.f) + 1e-5f);
  }
  __syncthreads();
  for (int idx = tid; idx < 512; idx += 256) {
    int s = idx >> 6;
    s_sl[idx] = (s_u[idx] - s_st[s]) * s_st[8 + s];   // ln(h)
  }
  __syncthreads();
  for (int o = tid; o < 1024; o += 256) {
    int s = o >> 7, hh = o & 127;
    const float* w = w1 + hh * 64;
    float a = b1[hh];
#pragma unroll 8
    for (int i = 0; i < 64; ++i) a += s_sl[s*64+i] * w[i];
    s_act[o] = fmaxf(a, 0.f);
  }
  __syncthreads();
  for (int idx = tid; idx < 512; idx += 256) {
    int s = idx >> 6, d = idx & 63;
    const float* w = w2 + d * 128;
    float o = b2[d];
#pragma unroll 8
    for (int i = 0; i < 128; ++i) o += s_act[s*128+i] * w[i];
    float val = s_sl[idx] + o;
    slots[(size_t)b * 512 + idx] = val;
    if (write_out) out[(size_t)b * 512 + idx] = val;
  }
}

// ---------------------------------------------------------------- launch
extern "C" void kernel_launch(void* const* d_in, const int* in_sizes, int n_in,
                              void* d_out, int out_size, void* d_ws, size_t ws_size,
                              hipStream_t stream) {
  const float* x      = (const float*)d_in[0];
  const float* noise  = (const float*)d_in[1];
  const float* fc_q_w = (const float*)d_in[2];
  const float* fc_q_b = (const float*)d_in[3];
  const float* fc_k_w = (const float*)d_in[4];
  const float* fc_k_b = (const float*)d_in[5];
  const float* fc_v_w = (const float*)d_in[6];
  const float* fc_v_b = (const float*)d_in[7];
  const float* w_ih   = (const float*)d_in[8];
  const float* w_hh   = (const float*)d_in[9];
  const float* b_ih   = (const float*)d_in[10];
  const float* b_hh   = (const float*)d_in[11];
  const float* w1     = (const float*)d_in[12];
  const float* b1     = (const float*)d_in[13];
  const float* w2     = (const float*)d_in[14];
  const float* b2     = (const float*)d_in[15];
  const float* mu     = (const float*)d_in[16];
  const float* sigma  = (const float*)d_in[17];

  char* ws = (char*)d_ws;                          // all offsets 4KB-aligned
  float* slots   = (float*)(ws + 0);               // 131072 B
  f16*   wq_h    = (f16*)  (ws + 131072);          // 8192 B
  f16*   wk_h    = (f16*)  (ws + 139264);          // 8192 B
  f16*   wv_h    = (f16*)  (ws + 147456);          // 8192 B
  f16*   q_h     = (f16*)  (ws + 155648);          // 131072 B  [b][16][64]
  float* colsum  = (float*)(ws + 286720);          // 4096 B    [b][16]
  float* upd_acc = (float*)(ws + 290816);          // 262144 B  [b][64][16]
  f16*   attn_t  = (f16*)  (ws + 552960);          // 8388608 B [b][16][4096]
  f16*   k_h     = (f16*)  (ws + 8941568);         // 33554432 B [b][4096][64]
  f16*   v_t     = (f16*)  (ws + 42496000);        // 33554432 B [b][64][4096]
  float* out = (float*)d_out;

  k_convert<<<16, 256, 0, stream>>>(fc_q_w, fc_k_w, fc_v_w, wq_h, wk_h, wv_h);
  k_init_slots<<<128, 256, 0, stream>>>(mu, sigma, noise, slots);
  k_ln_proj<<<2048, 256, 0, stream>>>(x, wk_h, wv_h, fc_k_b, fc_v_b, k_h, v_t);
  for (int it = 0; it < 3; ++it) {
    k_q<<<64, 32, 0, stream>>>(slots, wq_h, fc_q_b, q_h, colsum, upd_acc);
    k_attn<<<2048, 256, 0, stream>>>(k_h, q_h, attn_t, colsum);
    k_updates<<<64, 256, 0, stream>>>(v_t, attn_t, upd_acc);
    k_gru<<<64, 256, 0, stream>>>(upd_acc, colsum, w_ih, w_hh, b_ih, b_hh,
                                  w1, b1, w2, b2, slots, out, (it == 2) ? 1 : 0);
  }
}
